// GraphSelfAttention_12816182411318
// MI455X (gfx1250) — matrix operand order repaired and checked by reading, not compiled, not run on hardware
//
#include <hip/hip_runtime.h>

#define B_  4
#define N_  2048
#define D_  512
#define H_  8
#define DH_ 64
#define BN_ (B_ * N_)
#define DD_ (D_ * D_)

typedef __attribute__((ext_vector_type(16))) __bf16 v16bf;
typedef __attribute__((ext_vector_type(8)))  __bf16 v8bf;
typedef __attribute__((ext_vector_type(8)))  float  v8f;
typedef __attribute__((ext_vector_type(4)))  unsigned int v4u;
typedef __attribute__((ext_vector_type(4)))  int v4i;
typedef __attribute__((ext_vector_type(8)))  int v8i;

#if __has_builtin(__builtin_amdgcn_tensor_load_to_lds)
#define USE_TDM 1
#else
#define USE_TDM 0
#endif

__device__ __forceinline__ __bf16 f2bf(float f) {
  unsigned int u = __builtin_bit_cast(unsigned int, f);
  unsigned int r = (u + 0x7FFFu + ((u >> 16) & 1u)) >> 16;
  unsigned short s = (unsigned short)r;
  return __builtin_bit_cast(__bf16, s);
}

__device__ __forceinline__ v8f zero8() {
  v8f z;
#pragma unroll
  for (int i = 0; i < 8; ++i) z[i] = 0.f;
  return z;
}

__device__ __forceinline__ v8f wmma_bf16(v16bf a, v16bf b, v8f c) {
  // 8 args: (neg_a, A, neg_b, B, c_mod, C, reuse_a, reuse_b)
  return __builtin_amdgcn_wmma_f32_16x16x32_bf16(false, a, false, b, (short)0, c,
                                                 false, false);
}

// A-matrix 16x32 bf16 fragment from row-major tile (row stride ld elements).
// Lane l: M = l%16 ; element e: K = (l/16)*16 + e.
__device__ __forceinline__ v16bf load_a16x32(const __bf16* tile, int ld) {
  const int lm = threadIdx.x & 15;
  const int lh = (threadIdx.x >> 4) & 1;
  const __bf16* p = tile + lm * ld;
  v16bf a = *(const v16bf*)(p + lh * 16);
  return a;
}

// B-matrix 32x16 bf16 fragment, source stored as src[n][k] row-major
// (i.e. B[k][n] = src[n*ld + k]). Lane l: N = l%16 ; K = (l/16)*16 + e.
__device__ __forceinline__ v16bf load_b32x16_t(const __bf16* tile, int ld) {
  const int lm = threadIdx.x & 15;
  const int lh = (threadIdx.x >> 4) & 1;
  return *(const v16bf*)(tile + lm * ld + lh * 16);
}

__device__ __forceinline__ float redmax16(float v) {
  v = fmaxf(v, __shfl_xor(v, 1, 32));
  v = fmaxf(v, __shfl_xor(v, 2, 32));
  v = fmaxf(v, __shfl_xor(v, 4, 32));
  v = fmaxf(v, __shfl_xor(v, 8, 32));
  return v;
}
__device__ __forceinline__ float redsum16(float v) {
  v += __shfl_xor(v, 1, 32);
  v += __shfl_xor(v, 2, 32);
  v += __shfl_xor(v, 4, 32);
  v += __shfl_xor(v, 8, 32);
  return v;
}

#if USE_TDM
// Issue a TDM 2-D tile load: global (bf16, row-major, stride0 elems/row)
// -> LDS at byte offset lds_off. Dims/strides in elements (data_size = 2B).
__device__ __forceinline__ void tdm_load_2d(unsigned lds_off, const void* gptr,
                                            unsigned tile_d0, unsigned tile_d1,
                                            unsigned tensor_d0,
                                            unsigned tensor_d1,
                                            unsigned stride0) {
  unsigned long long ga = (unsigned long long)gptr;
  v4u g0;
  g0[0] = 1u;                                    // count=1, user descriptor
  g0[1] = lds_off;                               // lds_addr (bytes)
  g0[2] = (unsigned)(ga & 0xffffffffu);          // global_addr[31:0]
  g0[3] = (unsigned)((ga >> 32) & 0x1ffffffu) | (2u << 30);  // [56:32]|type=2
  v8i g1;
  g1[0] = (int)(1u << 16);                       // data_size=1 (2 bytes)
  g1[1] = (int)((tensor_d0 & 0xffffu) << 16);    // tensor_dim0[15:0]
  g1[2] = (int)(((tensor_d0 >> 16) & 0xffffu) | ((tensor_d1 & 0xffffu) << 16));
  g1[3] = (int)(((tensor_d1 >> 16) & 0xffffu) | ((tile_d0 & 0xffffu) << 16));
  g1[4] = (int)(tile_d1 & 0xffffu);              // tile_dim1 (tile_dim2 = 0)
  g1[5] = (int)stride0;                          // tensor_dim0_stride[31:0]
  g1[6] = 0;
  g1[7] = 0;
  v4i z4;
#pragma unroll
  for (int i = 0; i < 4; ++i) z4[i] = 0;
#if defined(__clang_major__) && (__clang_major__ >= 23)
  v8i z8;
#pragma unroll
  for (int i = 0; i < 8; ++i) z8[i] = 0;
  __builtin_amdgcn_tensor_load_to_lds(g0, g1, z4, z4, z8, 0);
#else
  __builtin_amdgcn_tensor_load_to_lds(g0, g1, z4, z4, 0);
#endif
}
#endif

// ---------------------------------------------------------------- conversion
__global__ void k_f32_to_bf16(const float* __restrict__ src,
                              __bf16* __restrict__ dst, int n) {
  int i = blockIdx.x * blockDim.x + threadIdx.x;
  int stride = gridDim.x * blockDim.x;
  for (; i < n; i += stride) dst[i] = f2bf(src[i]);
}

// ------------------------------------------- fused QKV projection + head-LN
// grid: (BN/128, H, 3[q,k,v]); block: 256 (8 waves), each wave = 16 rows x 64
// cols (one head) of one projection. LN done in-register on C/D fragments.
__global__ void __launch_bounds__(256)
k_qkv_ln(const __bf16* __restrict__ xb,
         const __bf16* __restrict__ wq, const __bf16* __restrict__ wk,
         const __bf16* __restrict__ wv,
         const float* __restrict__ bq, const float* __restrict__ bk,
         const float* __restrict__ bv,
         const float* __restrict__ qg, const float* __restrict__ qbt,
         const float* __restrict__ kg, const float* __restrict__ kbt,
         __bf16* __restrict__ qn, __bf16* __restrict__ kn,
         __bf16* __restrict__ vt) {
  const int wave = threadIdx.x >> 5;
  const int lm = threadIdx.x & 15;
  const int lh = (threadIdx.x >> 4) & 1;
  const int rt = blockIdx.x * 8 + wave;  // 16-row tile index over B*N
  const int h = blockIdx.y;
  const int pj = blockIdx.z;  // 0=Q 1=K 2=V
  const __bf16* W = (pj == 0) ? wq : (pj == 1) ? wk : wv;
  const float* bias = (pj == 0) ? bq : (pj == 1) ? bk : bv;

  const __bf16* xt = xb + rt * 16 * D_;
  v8f acc[4];
#pragma unroll
  for (int t = 0; t < 4; ++t) acc[t] = zero8();

  for (int ks = 0; ks < D_ / 32; ++ks) {
    v16bf a = load_a16x32(xt + ks * 32, D_);
#pragma unroll
    for (int t = 0; t < 4; ++t) {
      v16bf b = load_b32x16_t(W + (h * DH_ + t * 16) * D_ + ks * 32, D_);
      acc[t] = wmma_bf16(a, b, acc[t]);
    }
  }
#pragma unroll
  for (int t = 0; t < 4; ++t) {
    float bb = bias[h * DH_ + t * 16 + lm];
#pragma unroll
    for (int r = 0; r < 8; ++r) acc[t][r] += bb;
  }

  const int grow0 = rt * 16;
  const int b = grow0 >> 11;       // / N_
  const int n0 = grow0 & (N_ - 1);
  const int bh = b * H_ + h;

  if (pj == 2) {
    // V stored transposed: (B,H,DH,N) so PV B-fragments are contiguous loads
#pragma unroll
    for (int t = 0; t < 4; ++t) {
      int d = t * 16 + lm;
#pragma unroll
      for (int r = 0; r < 8; ++r) {
        int n = n0 + r + 8 * lh;
        vt[(bh * DH_ + d) * N_ + n] = f2bf(acc[t][r]);
      }
    }
  } else {
    const float* g = (pj == 0) ? qg : kg;
    const float* be = (pj == 0) ? qbt : kbt;
    __bf16* outp = (pj == 0) ? qn : kn;
#pragma unroll
    for (int r = 0; r < 8; ++r) {
      float s = 0.f, sq = 0.f;
#pragma unroll
      for (int t = 0; t < 4; ++t) { float v = acc[t][r]; s += v; sq += v * v; }
      s = redsum16(s);
      sq = redsum16(sq);
      float mu = s * (1.f / DH_);
      float var = sq * (1.f / DH_) - mu * mu;
      float rstd = rsqrtf(var + 1e-6f);
      int n = n0 + r + 8 * lh;
#pragma unroll
      for (int t = 0; t < 4; ++t) {
        int d = t * 16 + lm;
        float v = (acc[t][r] - mu) * rstd * g[d] + be[d];
        outp[(bh * N_ + n) * DH_ + d] = f2bf(v);
      }
    }
  }
}

// ---------------------------------------------------- flash attention core
// grid: (N/128, H, B); block 256 (8 waves). The block shares double-buffered
// K / V^T tiles in LDS (staged by the Tensor Data Mover when available);
// each wave owns 16 query rows and runs online softmax over key blocks of 32.
__global__ void __launch_bounds__(256)
k_flash(const __bf16* __restrict__ qn, const __bf16* __restrict__ kn,
        const __bf16* __restrict__ vt, const float* __restrict__ bias,
        __bf16* __restrict__ om) {
  __shared__ __align__(64) __bf16 ktile[2][32][64];  // keys x d   (8 KB)
  __shared__ __align__(64) __bf16 vtile[2][64][32];  // d x keys   (8 KB)
  __shared__ __align__(64) __bf16 lp[8][16][32];     // per-wave P (8 KB)

  const int wave = threadIdx.x >> 5;
  const int lm = threadIdx.x & 15;
  const int lh = (threadIdx.x >> 4) & 1;
  const int h = blockIdx.y;
  const int b = blockIdx.z;
  const int bh = b * H_ + h;
  const int q0 = (blockIdx.x * 8 + wave) * 16;

  const __bf16* qbase = qn + (bh * N_ + q0) * DH_;
  v16bf aq0 = load_a16x32(qbase, DH_);       // d = 0..31
  v16bf aq1 = load_a16x32(qbase + 32, DH_);  // d = 32..63

  v8f o[4];
#pragma unroll
  for (int t = 0; t < 4; ++t) o[t] = zero8();
  float rmax[8], lsum[8];
#pragma unroll
  for (int r = 0; r < 8; ++r) { rmax[r] = -1e30f; lsum[r] = 0.f; }

  const float* brow = bias + (b * N_ + q0) * N_;
  __bf16* lpw = &lp[wave][0][0];

  // ---- K/V tile staging into LDS buffer nb ----
  auto stage = [&](int kb2, int nb) {
#if USE_TDM
    if (threadIdx.x < 32) {  // one wave issues both DMAs (EXEC-independent)
      tdm_load_2d((unsigned)(size_t)(&ktile[nb][0][0]),
                  kn + (bh * N_ + kb2) * DH_,
                  /*tile*/ 64, 32, /*tensor*/ 64, N_, /*stride0*/ 64);
      tdm_load_2d((unsigned)(size_t)(&vtile[nb][0][0]),
                  vt + (size_t)(bh * DH_) * N_ + kb2,
                  /*tile*/ 32, 64, /*tensor*/ N_, DH_, /*stride0*/ N_);
    }
#else
    // cooperative fallback: 256 threads x 16B for each 4 KB tile
    const int t = threadIdx.x;
    *(v8bf*)(&ktile[nb][0][0] + t * 8) =
        *(const v8bf*)(kn + (bh * N_ + kb2) * DH_ + t * 8);
    const int row = t >> 2, seg = t & 3;
    *(v8bf*)(&vtile[nb][row][seg * 8]) =
        *(const v8bf*)(vt + (bh * DH_ + row) * N_ + kb2 + seg * 8);
#endif
  };

  stage(0, 0);
#if USE_TDM
  if (threadIdx.x < 32) __builtin_amdgcn_s_wait_tensorcnt(0);
#endif
  __syncthreads();

  int buf = 0;
  for (int kb = 0; kb < N_; kb += 32, buf ^= 1) {
    if (kb + 32 < N_) stage(kb + 32, buf ^ 1);  // prefetch next block
    __builtin_prefetch(brow + kb + 32, 0, 1);   // bias stream (HBM)

    const __bf16* kt = &ktile[buf][0][0];
    const __bf16* vl = &vtile[buf][0][0];

    // scores: two 16x16 tiles over keys kb..kb+31, K-dim = DH = 64
    v8f s0 = zero8(), s1 = zero8();
    s0 = wmma_bf16(aq0, load_b32x16_t(kt, 64), s0);
    s0 = wmma_bf16(aq1, load_b32x16_t(kt + 32, 64), s0);
    s1 = wmma_bf16(aq0, load_b32x16_t(kt + 16 * 64, 64), s1);
    s1 = wmma_bf16(aq1, load_b32x16_t(kt + 16 * 64 + 32, 64), s1);

#pragma unroll
    for (int r = 0; r < 8; ++r) {
      const float* bp = brow + (r + 8 * lh) * N_ + kb;
      float v0 = s0[r] * 0.125f + bp[lm];  // 1/sqrt(64)
      float v1 = s1[r] * 0.125f + bp[16 + lm];
      float m = redmax16(fmaxf(v0, v1));
      float nm = fmaxf(rmax[r], m);
      float corr = __expf(rmax[r] - nm);
      rmax[r] = nm;
      float p0 = __expf(v0 - nm);
      float p1 = __expf(v1 - nm);
      lsum[r] = lsum[r] * corr + redsum16(p0 + p1);
#pragma unroll
      for (int t = 0; t < 4; ++t) o[t][r] *= corr;
      // C/D layout -> LDS (row = r + 8*(lane/16), col = lane%16 [+16])
      lpw[(r + 8 * lh) * 32 + lm] = f2bf(p0);
      lpw[(r + 8 * lh) * 32 + 16 + lm] = f2bf(p1);
    }

    // re-read P as an A 16x32 fragment (same wave; dscnt ordering by compiler)
    v16bf pa = load_a16x32(lpw, 32);
#pragma unroll
    for (int t = 0; t < 4; ++t) {
      o[t] = wmma_bf16(pa, load_b32x16_t(vl + t * 16 * 32, 32), o[t]);
    }

#if USE_TDM
    if (threadIdx.x < 32) __builtin_amdgcn_s_wait_tensorcnt(0);
#endif
    __syncthreads();  // next buffer ready; everyone done with current buffer
  }

  // epilogue: normalize and write merged-head layout (B, N, H*DH)
#pragma unroll
  for (int r = 0; r < 8; ++r) {
    float inv = 1.f / lsum[r];
    int q = q0 + r + 8 * lh;
#pragma unroll
    for (int t = 0; t < 4; ++t) {
      om[(b * N_ + q) * D_ + h * DH_ + t * 16 + lm] = f2bf(o[t][r] * inv);
    }
  }
}

// ------------------------------------------------------- output projection
// grid: (BN/128, D/64); block 256 (8 waves): wave = 16 rows x 64 cols.
__global__ void __launch_bounds__(256)
k_outproj(const __bf16* __restrict__ om, const __bf16* __restrict__ wo,
          const float* __restrict__ bo, float* __restrict__ out) {
  const int wave = threadIdx.x >> 5;
  const int lm = threadIdx.x & 15;
  const int lh = (threadIdx.x >> 4) & 1;
  const int rt = blockIdx.x * 8 + wave;
  const int c0 = blockIdx.y * 64;
  const __bf16* at = om + rt * 16 * D_;

  v8f acc[4];
#pragma unroll
  for (int t = 0; t < 4; ++t) acc[t] = zero8();

  for (int ks = 0; ks < D_ / 32; ++ks) {
    v16bf a = load_a16x32(at + ks * 32, D_);
#pragma unroll
    for (int t = 0; t < 4; ++t) {
      v16bf bf = load_b32x16_t(wo + (c0 + t * 16) * D_ + ks * 32, D_);
      acc[t] = wmma_bf16(a, bf, acc[t]);
    }
  }
#pragma unroll
  for (int t = 0; t < 4; ++t) {
    float bb = bo[c0 + t * 16 + lm];
#pragma unroll
    for (int r = 0; r < 8; ++r) {
      out[(rt * 16 + r + 8 * lh) * D_ + c0 + t * 16 + lm] = acc[t][r] + bb;
    }
  }
}

// --------------------------------------------------------------------------
extern "C" void kernel_launch(void* const* d_in, const int* in_sizes, int n_in,
                              void* d_out, int out_size, void* d_ws,
                              size_t ws_size, hipStream_t stream) {
  const float* x   = (const float*)d_in[0];
  const float* ab  = (const float*)d_in[1];
  const float* Wq  = (const float*)d_in[2];
  const float* bq  = (const float*)d_in[3];
  const float* Wk  = (const float*)d_in[4];
  const float* bk  = (const float*)d_in[5];
  const float* Wv  = (const float*)d_in[6];
  const float* bv  = (const float*)d_in[7];
  const float* Wo  = (const float*)d_in[8];
  const float* bo  = (const float*)d_in[9];
  const float* qg  = (const float*)d_in[10];
  const float* qbt = (const float*)d_in[11];
  const float* kg  = (const float*)d_in[12];
  const float* kbt = (const float*)d_in[13];
  float* out = (float*)d_out;

  // bf16 workspace carve-out (~44 MB; fits global L2)
  __bf16* xb  = (__bf16*)d_ws;                 // (B*N, D)
  __bf16* wqb = xb + (size_t)BN_ * D_;
  __bf16* wkb = wqb + DD_;
  __bf16* wvb = wkb + DD_;
  __bf16* wob = wvb + DD_;
  __bf16* qnb = wob + DD_;                     // (B,H,N,DH) normalized Q
  __bf16* knb = qnb + (size_t)BN_ * D_;        // (B,H,N,DH) normalized K
  __bf16* vtb = knb + (size_t)BN_ * D_;        // (B,H,DH,N) V transposed
  __bf16* omb = vtb + (size_t)BN_ * D_;        // (B,N,D) attn output

  k_f32_to_bf16<<<512, 256, 0, stream>>>(x, xb, BN_ * D_);
  k_f32_to_bf16<<<128, 256, 0, stream>>>(Wq, wqb, DD_);
  k_f32_to_bf16<<<128, 256, 0, stream>>>(Wk, wkb, DD_);
  k_f32_to_bf16<<<128, 256, 0, stream>>>(Wv, wvb, DD_);
  k_f32_to_bf16<<<128, 256, 0, stream>>>(Wo, wob, DD_);

  k_qkv_ln<<<dim3(BN_ / 128, H_, 3), 256, 0, stream>>>(
      xb, wqb, wkb, wvb, bq, bk, bv, qg, qbt, kg, kbt, qnb, knb, vtb);

  k_flash<<<dim3(N_ / 128, H_, B_), 256, 0, stream>>>(qnb, knb, vtb, ab, omb);

  k_outproj<<<dim3(BN_ / 128, D_ / 64), 256, 0, stream>>>(omb, wob, bo, out);
}
